// ls2ic_64304250355948
// MI455X (gfx1250) — compile-verified, matrix operand order checked by reading, not run on hardware
//
#include <hip/hip_runtime.h>
#include <hip/hip_bf16.h>

typedef __attribute__((ext_vector_type(16))) _Float16 v16h;
typedef __attribute__((ext_vector_type(8)))  float    v8f;

#define BSZ 1024
#define NAG 64
#define NLK 32
#define NFE 16
#define HID 32
#define MSGD 32
#define ACT 16

static __device__ __forceinline__ v8f wmma_f16(v16h a, v16h b, v8f c) {
  return __builtin_amdgcn_wmma_f32_16x16x32_f16(false, a, false, b, (short)0, c, false, false);
}

// A fragment (16x32 f16) from row-major f16 LDS, zero-padded K (ISA 7.12.2 layout:
// lanes 0-15: K 0..7 & 16..23, lanes 16-31: K 8..15 & 24..31, m = lane&15).
static __device__ __forceinline__ v16h frag_a_h(const _Float16* s, int ld, int row0,
                                                int k0, int kval, int lane) {
  const int m  = row0 + (lane & 15);
  const int kg = (lane >> 4) << 3;
  v16h a;
#pragma unroll
  for (int i = 0; i < 8; ++i) {
    const int kA = kg + i, kB = 16 + kg + i;
    a[i]     = (kA < kval) ? s[m * ld + k0 + kA] : (_Float16)0.0f;
    a[i + 8] = (kB < kval) ? s[m * ld + k0 + kB] : (_Float16)0.0f;
  }
  return a;
}

// A fragment from row-major f32 LDS (states tile / probs / h_prev), zero-padded K.
static __device__ __forceinline__ v16h frag_a_f(const float* s, int ld, int row0,
                                                int k0, int kval, int lane) {
  const int m  = row0 + (lane & 15);
  const int kg = (lane >> 4) << 3;
  v16h a;
#pragma unroll
  for (int i = 0; i < 8; ++i) {
    const int kA = kg + i, kB = 16 + kg + i;
    a[i]     = (kA < kval) ? (_Float16)s[m * ld + k0 + kA] : (_Float16)0.0f;
    a[i + 8] = (kB < kval) ? (_Float16)s[m * ld + k0 + kB] : (_Float16)0.0f;
  }
  return a;
}

// B fragment (32x16 f16) from row-major f32 global weight [K x N], zero-padded K.
static __device__ __forceinline__ v16h frag_b_g(const float* s, int ld, int k0,
                                                int n0, int kval, int lane) {
  const int n  = n0 + (lane & 15);
  const int kg = (lane >> 4) << 3;
  v16h b;
#pragma unroll
  for (int i = 0; i < 8; ++i) {
    const int kA = kg + i, kB = 16 + kg + i;
    b[i]     = (kA < kval) ? (_Float16)s[(k0 + kA) * ld + n] : (_Float16)0.0f;
    b[i + 8] = (kB < kval) ? (_Float16)s[(k0 + kB) * ld + n] : (_Float16)0.0f;
  }
  return b;
}

// B = Kmat^T : b(k,n) = Kmat[n0+n][cb+k], zero-padded K (scores B operand).
static __device__ __forceinline__ v16h frag_bT_h(const _Float16* s, int ld, int n0,
                                                 int cb, int kval, int lane) {
  const int n  = n0 + (lane & 15);
  const int kg = (lane >> 4) << 3;
  v16h b;
#pragma unroll
  for (int i = 0; i < 8; ++i) {
    const int kA = kg + i, kB = 16 + kg + i;
    b[i]     = (kA < kval) ? s[n * ld + cb + kA] : (_Float16)0.0f;
    b[i + 8] = (kB < kval) ? s[n * ld + cb + kB] : (_Float16)0.0f;
  }
  return b;
}

// B for P@V_head: 8 valid columns replicated into 16 (extra D columns discarded).
static __device__ __forceinline__ v16h frag_b_v8(const _Float16* s, int ld, int cb, int lane) {
  const int n  = lane & 7;
  const int kg = (lane >> 4) << 3;
  v16h b;
#pragma unroll
  for (int i = 0; i < 8; ++i) {
    b[i]     = s[(kg + i) * ld + cb + n];
    b[i + 8] = s[(16 + kg + i) * ld + cb + n];
  }
  return b;
}

// Store D fragment + scalar bias, optional relu, as f16 to row-major LDS.
static __device__ __forceinline__ void st_d_h(v8f c, _Float16* d, int ld, int row0, int col0,
                                              float bv, int relu, int lane) {
  const int n  = lane & 15;
  const int mb = (lane >> 4) << 3;
#pragma unroll
  for (int v = 0; v < 8; ++v) {
    float x = c[v] + bv;
    if (relu) x = fmaxf(x, 0.0f);
    d[(row0 + mb + v) * ld + col0 + n] = (_Float16)x;
  }
}

// Store D fragment + scalar bias as f32 to row-major LDS.
static __device__ __forceinline__ void st_d_f(v8f c, float* d, int ld, int row0, int col0,
                                              float bv, int lane) {
  const int n  = lane & 15;
  const int mb = (lane >> 4) << 3;
#pragma unroll
  for (int v = 0; v < 8; ++v)
    d[(row0 + mb + v) * ld + col0 + n] = c[v] + bv;
}

// Store D fragment * scale (no bias) as f32 to row-major LDS (attention scores).
static __device__ __forceinline__ void st_d_sc(v8f c, float* d, int ld, int row0, int col0,
                                               float scale, int lane) {
  const int n  = lane & 15;
  const int mb = (lane >> 4) << 3;
#pragma unroll
  for (int v = 0; v < 8; ++v)
    d[(row0 + mb + v) * ld + col0 + n] = c[v] * scale;
}

// ---------------- Kernel A: link attention + encoder + GRU + messages ----------------
__global__ __launch_bounds__(64) void
fused_encode_gru_msg(const float* __restrict__ states, const float* __restrict__ hprev,
                     const float* __restrict__ wq,  const float* __restrict__ bq,
                     const float* __restrict__ wkv, const float* __restrict__ bkv,
                     const float* __restrict__ wo,  const float* __restrict__ bo,
                     const float* __restrict__ ln1g, const float* __restrict__ ln1b,
                     const float* __restrict__ ilw, const float* __restrict__ ilb,
                     const float* __restrict__ wih, const float* __restrict__ whh,
                     const float* __restrict__ bih, const float* __restrict__ bhh,
                     const float* __restrict__ msw, const float* __restrict__ msb,
                     const float* __restrict__ mgw, const float* __restrict__ mgb,
                     float* __restrict__ out_h, float* __restrict__ out_mm)
{
  __shared__ __align__(16) float sS[2][NLK][NFE]; // staged states tile (f32, async DMA)
  __shared__ _Float16 sQ[2][NLK][32];
  __shared__ _Float16 sK[2][NLK][32];
  __shared__ _Float16 sV[2][NLK][32];
  __shared__ float    sP[2][NLK][32];       // scores/probs, later reused [32][16] f32 proj
  __shared__ _Float16 sO[2][NLK][32];       // attention output (pre-proj)
  __shared__ _Float16 aft[16][NLK * NFE];   // 16 rows x 512 (post-LN, encoder A operand)
  __shared__ _Float16 encH[16][HID];
  __shared__ __align__(16) float hpF[16][HID]; // staged h_prev (f32, async DMA)
  __shared__ float    gi[16][96];
  __shared__ float    gh[16][96];
  __shared__ _Float16 hH[16][HID];
  __shared__ _Float16 membH[16][HID];

  const int agent = blockIdx.x & (NAG - 1);
  const int chunk = blockIdx.x >> 6;
  const int w     = threadIdx.x >> 5;
  const int lane  = threadIdx.x & 31;
  const int nn    = lane & 15;

  // Loop-invariant attention weight fragments + per-lane bias scalars (hoisted:
  // reused 8x per wave; keeps the hot loop free of weight reloads/converts).
  v16h bWq[2], bWkv[4];
#pragma unroll
  for (int nt = 0; nt < 2; ++nt) bWq[nt]  = frag_b_g(wq, 32, 0, nt * 16, NFE, lane);
#pragma unroll
  for (int nt = 0; nt < 4; ++nt) bWkv[nt] = frag_b_g(wkv, 64, 0, nt * 16, NFE, lane);
  const v16h bWo = frag_b_g(wo, NFE, 0, 0, 32, lane);
  float bqv[2], bkvv[4];
#pragma unroll
  for (int nt = 0; nt < 2; ++nt) bqv[nt]  = bq[nt * 16 + nn];
#pragma unroll
  for (int nt = 0; nt < 4; ++nt) bkvv[nt] = bkv[nt * 16 + nn];
  const float bov = bo[nn];
  float g1[NFE], b1[NFE];
#pragma unroll
  for (int f = 0; f < NFE; ++f) { g1[f] = ln1g[f]; b1[f] = ln1b[f]; }

  // -------- phase 1: per-pair link self-attention, 8 pairs per wave --------
#pragma unroll 1
  for (int i = 0; i < 8; ++i) {
    const int r = w * 8 + i;
    const int b = chunk * 16 + r;
    const size_t sbase = (size_t)(b * NAG + agent) * (NLK * NFE);
    {
      // Stage the 2KB states tile straight into LDS via the CDNA5 async data path
      // (GLOBAL_LOAD_ASYNC_TO_LDS_B128, tracked by ASYNCcnt): 4 issues x 32 lanes x 16B.
#if defined(__gfx1250__)
      const unsigned ldsBase = (unsigned)(uintptr_t)&sS[w][0][0];
      const char* gBase = (const char*)(states + sbase);
#pragma unroll
      for (int t = 0; t < 4; ++t) {
        const unsigned ldsOff = ldsBase + (unsigned)(lane * 16 + t * 512);
        const char* ga = gBase + lane * 16 + t * 512;
        asm volatile("global_load_async_to_lds_b128 %0, %1, off"
                     :: "v"(ldsOff), "v"(ga) : "memory");
      }
      asm volatile("s_wait_asynccnt 0x0" ::: "memory");
#else
      const float* src = states + sbase + (size_t)lane * NFE;
#pragma unroll
      for (int f = 0; f < NFE; ++f) sS[w][lane][f] = src[f];
#endif
      if (i < 7) __builtin_prefetch(states + sbase + (size_t)NAG * NLK * NFE, 0, 1);
    }
    __syncthreads();

    // Q = S@Wq+bq ; K,V = S@Wkv+bkv  (K dim = 16, zero-padded to 32)
#pragma unroll
    for (int mt = 0; mt < 2; ++mt) {
      v16h aS = frag_a_f(&sS[w][0][0], NFE, mt * 16, 0, NFE, lane);
#pragma unroll
      for (int nt = 0; nt < 2; ++nt) {
        v8f c = {};
        c = wmma_f16(aS, bWq[nt], c);
        st_d_h(c, &sQ[w][0][0], 32, mt * 16, nt * 16, bqv[nt], 0, lane);
      }
#pragma unroll
      for (int nt = 0; nt < 4; ++nt) {
        v8f c = {};
        c = wmma_f16(aS, bWkv[nt], c);
        if (nt < 2) st_d_h(c, &sK[w][0][0], 32, mt * 16, nt * 16,       bkvv[nt], 0, lane);
        else        st_d_h(c, &sV[w][0][0], 32, mt * 16, (nt - 2) * 16, bkvv[nt], 0, lane);
      }
    }
    __syncthreads();

    // 4 heads, head dim 8
#pragma unroll 1
    for (int h = 0; h < 4; ++h) {
      const int cb = h * 8;
#pragma unroll
      for (int mt = 0; mt < 2; ++mt) {
        v16h aQ = frag_a_h(&sQ[w][0][0], 32, mt * 16, cb, 8, lane);
#pragma unroll
        for (int nt = 0; nt < 2; ++nt) {
          v8f c = {};
          c = wmma_f16(aQ, frag_bT_h(&sK[w][0][0], 32, nt * 16, cb, 8, lane), c);
          st_d_sc(c, &sP[w][0][0], 32, mt * 16, nt * 16, 0.35355339059f, lane);
        }
      }
      __syncthreads();
      {                                    // softmax: one lane per query row
        float mx = -1e30f;
#pragma unroll
        for (int j = 0; j < 32; ++j) mx = fmaxf(mx, sP[w][lane][j]);
        float sum = 0.0f;
#pragma unroll
        for (int j = 0; j < 32; ++j) { float e = __expf(sP[w][lane][j] - mx); sP[w][lane][j] = e; sum += e; }
        const float rs = 1.0f / sum;
#pragma unroll
        for (int j = 0; j < 32; ++j) sP[w][lane][j] *= rs;
      }
      __syncthreads();
#pragma unroll
      for (int mt = 0; mt < 2; ++mt) {     // O_h = P @ V_h
        v16h aP = frag_a_f(&sP[w][0][0], 32, mt * 16, 0, 32, lane);
        v8f c = {};
        c = wmma_f16(aP, frag_b_v8(&sV[w][0][0], 32, cb, lane), c);
        const int mb = (lane >> 4) << 3;
        if (nn < 8) {
#pragma unroll
          for (int v = 0; v < 8; ++v) sO[w][mt * 16 + mb + v][cb + nn] = (_Float16)c[v];
        }
      }
      __syncthreads();
    }

    // out-projection (reuse sP[w] as [32][16] f32)
    float* proj = &sP[w][0][0];
#pragma unroll
    for (int mt = 0; mt < 2; ++mt) {
      v16h aO = frag_a_h(&sO[w][0][0], 32, mt * 16, 0, 32, lane);
      v8f c = {};
      c = wmma_f16(aO, bWo, c);
      st_d_f(c, proj, NFE, mt * 16, 0, bov, lane);
    }
    __syncthreads();

    {                                      // residual + LayerNorm(nf=16), lane = link row
      float x[NFE];
      float mean = 0.0f;
#pragma unroll
      for (int f = 0; f < NFE; ++f) { x[f] = proj[lane * NFE + f] + sS[w][lane][f]; mean += x[f]; }
      mean *= (1.0f / NFE);
      float var = 0.0f;
#pragma unroll
      for (int f = 0; f < NFE; ++f) { float d = x[f] - mean; var += d * d; }
      var *= (1.0f / NFE);
      const float rstd = rsqrtf(var + 1e-5f);
#pragma unroll
      for (int f = 0; f < NFE; ++f)
        aft[r][lane * NFE + f] = (_Float16)((x[f] - mean) * rstd * g1[f] + b1[f]);
    }
    __syncthreads();
  }

  // -------- phase 2: encoder GEMM 16x512x32, each wave one 16-col tile --------
  {
    v8f c = {};
    for (int kc = 0; kc < 16; ++kc) {
      v16h aA = frag_a_h(&aft[0][0], NLK * NFE, 0, kc * 32, 32, lane);
      c = wmma_f16(aA, frag_b_g(ilw, HID, kc * 32, w * 16, 32, lane), c);
    }
    st_d_h(c, &encH[0][0], HID, 0, w * 16, ilb[w * 16 + nn], 0, lane);
  }
  {
    // Stage h_prev (16 rows x 128B, row stride NAG*HID*4B) via async-to-LDS:
    // each wave covers 8 rows with 2 issues (lane -> row = lane/8, 16B part = lane%8).
#if defined(__gfx1250__)
    const unsigned ldsBase = (unsigned)(uintptr_t)&hpF[0][0];
#pragma unroll
    for (int t = 0; t < 2; ++t) {
      const int row  = w * 8 + t * 4 + (lane >> 3);
      const int part = lane & 7;
      const unsigned ldsOff = ldsBase + (unsigned)(row * 128 + part * 16);
      const char* ga = (const char*)(hprev + (size_t)((chunk * 16 + row) * NAG + agent) * HID)
                       + part * 16;
      asm volatile("global_load_async_to_lds_b128 %0, %1, off"
                   :: "v"(ldsOff), "v"(ga) : "memory");
    }
    asm volatile("s_wait_asynccnt 0x0" ::: "memory");
#else
#pragma unroll
    for (int e = 0; e < 8; ++e) {
      const int idx = threadIdx.x * 8 + e, row = idx >> 5, col = idx & 31;
      hpF[row][col] = hprev[(size_t)((chunk * 16 + row) * NAG + agent) * HID + col];
    }
#endif
  }
  __syncthreads();

  // GRU gates: wave0 -> gi = enc@Wih+bih ; wave1 -> gh = hprev@Whh+bhh
  {
    const float* W  = (w == 0) ? wih : whh;
    const float* Bv = (w == 0) ? bih : bhh;
    float*       G  = (w == 0) ? &gi[0][0] : &gh[0][0];
    v16h aG = (w == 0) ? frag_a_h(&encH[0][0], HID, 0, 0, 32, lane)
                       : frag_a_f(&hpF[0][0],  HID, 0, 0, 32, lane);
#pragma unroll
    for (int nt = 0; nt < 6; ++nt) {
      v8f c = {};
      c = wmma_f16(aG, frag_b_g(W, 96, 0, nt * 16, 32, lane), c);
      st_d_f(c, G, 96, 0, nt * 16, Bv[nt * 16 + nn], lane);
    }
  }
  __syncthreads();

#pragma unroll
  for (int e = 0; e < 8; ++e) {            // GRU elementwise, write h
    const int idx = threadIdx.x * 8 + e, row = idx >> 5, col = idx & 31;
    const float ir = gi[row][col], iz = gi[row][32 + col], ig = gi[row][64 + col];
    const float hr = gh[row][col], hz = gh[row][32 + col], hg = gh[row][64 + col];
    const float rr = 1.0f / (1.0f + __expf(-(ir + hr)));
    const float zz = 1.0f / (1.0f + __expf(-(iz + hz)));
    const float nv = tanhf(ig + rr * hg);
    const float hp = hpF[row][col];
    const float hv = (1.0f - zz) * nv + zz * hp;
    hH[row][col] = (_Float16)hv;
    out_h[(size_t)((chunk * 16 + row) * NAG + agent) * HID + col] = hv;
  }
  __syncthreads();

  {                                        // memb = relu(h@ms_w+ms_b), one tile per wave
    v16h aH = frag_a_h(&hH[0][0], HID, 0, 0, 32, lane);
    v8f c = {};
    c = wmma_f16(aH, frag_b_g(msw, HID, 0, w * 16, 32, lane), c);
    st_d_h(c, &membH[0][0], HID, 0, w * 16, msb[w * 16 + nn], 1, lane);
  }
  __syncthreads();

  {                                        // messages = memb @ mg_w[agent] + mg_b[agent]
    v16h aM = frag_a_h(&membH[0][0], HID, 0, 0, 32, lane);
    v8f c = {};
    c = wmma_f16(aM, frag_b_g(mgw + (size_t)agent * HID * MSGD, MSGD, 0, w * 16, 32, lane), c);
    const int mb = (lane >> 4) << 3;
    const float bmv = mgb[agent * MSGD + w * 16 + nn];
#pragma unroll
    for (int v = 0; v < 8; ++v) {
      const int row = mb + v;
      out_mm[(size_t)((chunk * 16 + row) * NAG + agent) * MSGD + w * 16 + nn] = c[v] + bmv;
    }
  }
}

// -------- Kernel B: per-batch agent mean, broadcast in place --------
__global__ __launch_bounds__(32) void mean_msg(float* __restrict__ out_mm)
{
  const int b = blockIdx.x;
  const int f = threadIdx.x;
  float s = 0.0f;
  for (int a = 0; a < NAG; ++a) s += out_mm[(size_t)(b * NAG + a) * MSGD + f];
  s *= (1.0f / NAG);
  for (int a = 0; a < NAG; ++a) out_mm[(size_t)(b * NAG + a) * MSGD + f] = s;
}

// -------- Kernel C: comms "attention" (softmax over 1 key == identity) + Q heads --------
__global__ __launch_bounds__(32) void
comm_q_head(const float* __restrict__ out_h_in, const float* __restrict__ out_mm_in,
            const float* __restrict__ mawkv, const float* __restrict__ mabkv,
            const float* __restrict__ mawo,  const float* __restrict__ mabo,
            const float* __restrict__ ln2g,  const float* __restrict__ ln2b,
            const float* __restrict__ qsw,   const float* __restrict__ qsb,
            const float* __restrict__ ahw,   const float* __restrict__ ahb,
            float* __restrict__ out_q)
{
  __shared__ _Float16 mmH[16][HID];
  __shared__ _Float16 vH[16][HID];
  __shared__ float    sAtt[16][HID];
  __shared__ _Float16 featH[16][2 * HID];
  __shared__ _Float16 qembH[16][HID];

  const int agent = blockIdx.x & (NAG - 1);
  const int chunk = blockIdx.x >> 6;
  const int lane  = threadIdx.x;
  const int nn    = lane & 15;

  for (int row = 0; row < 16; ++row)
    mmH[row][lane] = (_Float16)out_mm_in[(size_t)((chunk * 16 + row) * NAG + agent) * MSGD + lane];
  __syncthreads();

#pragma unroll
  for (int nt = 0; nt < 2; ++nt) {         // V = mm @ ma_wkv[:, H:2H] + bkv[H:]
    v16h aM = frag_a_h(&mmH[0][0], HID, 0, 0, 32, lane);
    v8f c = {};
    c = wmma_f16(aM, frag_b_g(mawkv, 64, 0, HID + nt * 16, 32, lane), c);
    st_d_h(c, &vH[0][0], HID, 0, nt * 16, mabkv[HID + nt * 16 + nn], 0, lane);
  }
  __syncthreads();
#pragma unroll
  for (int nt = 0; nt < 2; ++nt) {         // attn out = V @ ma_wo + bo
    v16h aV = frag_a_h(&vH[0][0], HID, 0, 0, 32, lane);
    v8f c = {};
    c = wmma_f16(aV, frag_b_g(mawo, HID, 0, nt * 16, 32, lane), c);
    st_d_f(c, &sAtt[0][0], HID, 0, nt * 16, mabo[nt * 16 + nn], lane);
  }
  __syncthreads();

  if (lane < 16) {                         // residual + LN(H=32), feat = [h || after]
    const int row = lane;
    float x[HID], hvv[HID];
    float mean = 0.0f;
    for (int col = 0; col < HID; ++col) {
      const float hv = out_h_in[(size_t)((chunk * 16 + row) * NAG + agent) * HID + col];
      hvv[col] = hv;
      x[col] = hv + sAtt[row][col];
      mean += x[col];
    }
    mean *= (1.0f / HID);
    float var = 0.0f;
    for (int col = 0; col < HID; ++col) { const float d = x[col] - mean; var += d * d; }
    var *= (1.0f / HID);
    const float rstd = rsqrtf(var + 1e-5f);
    for (int col = 0; col < HID; ++col) {
      featH[row][col]       = (_Float16)hvv[col];
      featH[row][HID + col] = (_Float16)((x[col] - mean) * rstd * ln2g[col] + ln2b[col]);
    }
  }
  __syncthreads();

#pragma unroll
  for (int nt = 0; nt < 2; ++nt) {         // qemb = relu(feat @ qs_w + qs_b), K=64
    v8f c = {};
#pragma unroll
    for (int kc = 0; kc < 2; ++kc) {
      v16h aF = frag_a_h(&featH[0][0], 2 * HID, 0, kc * 32, 32, lane);
      c = wmma_f16(aF, frag_b_g(qsw, HID, kc * 32, nt * 16, 32, lane), c);
    }
    st_d_h(c, &qembH[0][0], HID, 0, nt * 16, qsb[nt * 16 + nn], 1, lane);
  }
  __syncthreads();

  {                                        // q = qemb @ ah_w[agent] + ah_b[agent]
    v16h aQ = frag_a_h(&qembH[0][0], HID, 0, 0, 32, lane);
    v8f c = {};
    c = wmma_f16(aQ, frag_b_g(ahw + (size_t)agent * HID * ACT, ACT, 0, 0, 32, lane), c);
    const int mb = (lane >> 4) << 3;
    const float bav = ahb[agent * ACT + nn];
#pragma unroll
    for (int v = 0; v < 8; ++v) {
      const int row = mb + v;
      out_q[(size_t)((chunk * 16 + row) * NAG + agent) * ACT + nn] = c[v] + bav;
    }
  }
}

extern "C" void kernel_launch(void* const* d_in, const int* in_sizes, int n_in,
                              void* d_out, int out_size, void* d_ws, size_t ws_size,
                              hipStream_t stream) {
  (void)in_sizes; (void)n_in; (void)out_size; (void)d_ws; (void)ws_size;
  const float* states  = (const float*)d_in[0];
  const float* hidden  = (const float*)d_in[1];
  const float* ia_wq   = (const float*)d_in[2];
  const float* ia_bq   = (const float*)d_in[3];
  const float* ia_wkv  = (const float*)d_in[4];
  const float* ia_bkv  = (const float*)d_in[5];
  const float* ia_wo   = (const float*)d_in[6];
  const float* ia_bo   = (const float*)d_in[7];
  const float* ln1_g   = (const float*)d_in[8];
  const float* ln1_b   = (const float*)d_in[9];
  const float* il_w    = (const float*)d_in[10];
  const float* il_b    = (const float*)d_in[11];
  const float* gru_wih = (const float*)d_in[12];
  const float* gru_whh = (const float*)d_in[13];
  const float* gru_bih = (const float*)d_in[14];
  const float* gru_bhh = (const float*)d_in[15];
  const float* ms_w    = (const float*)d_in[16];
  const float* ms_b    = (const float*)d_in[17];
  const float* mg_w    = (const float*)d_in[18];
  const float* mg_b    = (const float*)d_in[19];
  // d_in[20], d_in[21] (ma_wq, ma_bq) are dead code: softmax over a single key is 1.
  const float* ma_wkv  = (const float*)d_in[22];
  const float* ma_bkv  = (const float*)d_in[23];
  const float* ma_wo   = (const float*)d_in[24];
  const float* ma_bo   = (const float*)d_in[25];
  const float* ln2_g   = (const float*)d_in[26];
  const float* ln2_b   = (const float*)d_in[27];
  const float* qs_w    = (const float*)d_in[28];
  const float* qs_b    = (const float*)d_in[29];
  const float* ah_w    = (const float*)d_in[30];
  const float* ah_b    = (const float*)d_in[31];

  float* out    = (float*)d_out;
  float* out_q  = out;
  float* out_h  = out + (size_t)BSZ * NAG * ACT;
  float* out_mm = out_h + (size_t)BSZ * NAG * HID;

  const dim3 grid(NAG * (BSZ / 16));
  fused_encode_gru_msg<<<grid, dim3(64), 0, stream>>>(
      states, hidden, ia_wq, ia_bq, ia_wkv, ia_bkv, ia_wo, ia_bo, ln1_g, ln1_b,
      il_w, il_b, gru_wih, gru_whh, gru_bih, gru_bhh, ms_w, ms_b, mg_w, mg_b,
      out_h, out_mm);
  mean_msg<<<dim3(BSZ), dim3(32), 0, stream>>>(out_mm);
  comm_q_head<<<grid, dim3(32), 0, stream>>>(
      out_h, out_mm, ma_wkv, ma_bkv, ma_wo, ma_bo, ln2_g, ln2_b,
      qs_w, qs_b, ah_w, ah_b, out_q);
}
